// VectorQuantizerEMA_26740466384914
// MI455X (gfx1250) — matrix operand order repaired and checked by reading, not compiled, not run on hardware
//
#include <hip/hip_runtime.h>
#include <cstddef>

typedef __attribute__((ext_vector_type(16))) __bf16 v16bf;
typedef __attribute__((ext_vector_type(8)))  float  v8f;
typedef __attribute__((ext_vector_type(4)))  float  f4;

#define N_ROWS   262144      // 64*4096
#define DIM      64
#define NCODES   512
#define QELEMS   ((size_t)N_ROWS * DIM)
#define BLOCKS   (N_ROWS / 256)   // 8 waves * 32 rows per block

// ---- prologue: bf16 codebook + folded per-code constants ----
// scen4[c] = sqrt(ema[c]) * ||e_c||^2 + 4     (the +4 keeps distances > 0)
// msc2[c]  = -2 * sqrt(ema[c])
__global__ void vq_prep_kernel(const float* __restrict__ emb,
                               const float* __restrict__ ema,
                               __bf16* __restrict__ embB,
                               float* __restrict__ scen4,
                               float* __restrict__ msc2) {
    int c = blockIdx.x * blockDim.x + threadIdx.x;
    if (c >= NCODES) return;
    float s = 0.0f;
    const float* rp = emb + c * DIM;
    __bf16* wp = embB + c * DIM;
    #pragma unroll
    for (int d = 0; d < DIM; ++d) {
        float v = rp[d];
        s += v * v;
        wp[d] = (__bf16)v;
    }
    float sc = sqrtf(ema[c]);
    scen4[c] = sc * s + 4.0f;
    msc2[c]  = -2.0f * sc;
}

// ---- main: WMMA distances + packed-key argmin + gather + loss ----
__global__ __launch_bounds__(256) void vq_main_kernel(
    const float*  __restrict__ x,       // [N_ROWS, 64] fp32
    const float*  __restrict__ emb,     // [512, 64]    fp32 (gather)
    const __bf16* __restrict__ embB,    // [512, 64]    bf16
    const float*  __restrict__ scen4,   // [512]
    const float*  __restrict__ msc2,    // [512]
    float* __restrict__ qout,           // [N_ROWS, 64]
    int*   __restrict__ idxOut,         // [N_ROWS]
    float* __restrict__ partials)       // [BLOCKS]
{
    __shared__ int   sIdx[8][32];
    __shared__ float sLoss[8];

    const int tid  = threadIdx.x;
    const int wave = tid >> 5;
    const int lane = tid & 31;
    const int l16  = lane & 15;
    const int hi   = lane >> 4;

    const long rowBase = ((long)blockIdx.x * 8 + wave) * 32;
    const long row0 = rowBase + l16;         // A-tile 0: rows rowBase..+15
    const long row1 = rowBase + 16 + l16;    // A-tile 1: rows rowBase+16..+31

    const int o = hi * 8;
    const int segB[4] = { o, o + 16, o + 32, o + 48 };

    // ---- nontemporal-load this lane's share of 2x16 rows ----
    const float* rp0 = x + row0 * DIM;
    const float* rp1 = x + row1 * DIM;
    f4 xv0[8], xv1[8];
    #pragma unroll
    for (int s2 = 0; s2 < 4; ++s2) {
        xv0[2*s2]   = __builtin_nontemporal_load((const f4*)(rp0 + segB[s2]));
        xv0[2*s2+1] = __builtin_nontemporal_load((const f4*)(rp0 + segB[s2] + 4));
        xv1[2*s2]   = __builtin_nontemporal_load((const f4*)(rp1 + segB[s2]));
        xv1[2*s2+1] = __builtin_nontemporal_load((const f4*)(rp1 + segB[s2] + 4));
    }

    // ---- row norms (lane l + lane l+16 hold one row) ----
    float p0 = 0.0f, p1 = 0.0f;
    #pragma unroll
    for (int i = 0; i < 8; ++i) {
        p0 += xv0[i].x*xv0[i].x + xv0[i].y*xv0[i].y + xv0[i].z*xv0[i].z + xv0[i].w*xv0[i].w;
        p1 += xv1[i].x*xv1[i].x + xv1[i].y*xv1[i].y + xv1[i].z*xv1[i].z + xv1[i].w*xv1[i].w;
    }
    float xn0 = p0 + __shfl_xor(p0, 16, 32);
    float xn1 = p1 + __shfl_xor(p1, 16, 32);

    // C-operand init: acc[j] starts at -0.5*||x_row(j)||^2, so after the two
    // WMMAs acc[j] = dot - 0.5*xn and d = fma(msc2, acc, scen4) is the scaled
    // distance (+4) in a single VALU op per entry.
    v8f c0, c1;
    #pragma unroll
    for (int j = 0; j < 8; ++j) {
        c0[j] = -0.5f * __shfl(xn0, j + 8 * hi, 32);
        c1[j] = -0.5f * __shfl(xn1, j + 8 * hi, 32);
    }

    // ---- bf16 A fragments (2 tiles x 2 K-chunks) ----
    float xs0[32], xs1[32];
    #pragma unroll
    for (int i = 0; i < 8; ++i) {
        xs0[4*i] = xv0[i].x; xs0[4*i+1] = xv0[i].y; xs0[4*i+2] = xv0[i].z; xs0[4*i+3] = xv0[i].w;
        xs1[4*i] = xv1[i].x; xs1[4*i+1] = xv1[i].y; xs1[4*i+2] = xv1[i].z; xs1[4*i+3] = xv1[i].w;
    }
    v16bf a00, a01, a10, a11;
    #pragma unroll
    for (int i = 0; i < 16; ++i) {
        a00[i] = (__bf16)xs0[i];  a01[i] = (__bf16)xs0[16 + i];
        a10[i] = (__bf16)xs1[i];  a11[i] = (__bf16)xs1[16 + i];
    }

    // ---- sweep 32 column tiles; packed-key running argmin ----
    unsigned key0[8], key1[8];
    #pragma unroll
    for (int j = 0; j < 8; ++j) { key0[j] = 0xFFFFFFFFu; key1[j] = 0xFFFFFFFFu; }

    for (int t = 0; t < 32; ++t) {
        const int code = t * 16 + l16;
        const __bf16* bp = embB + code * DIM + hi * 8;    // 16B-aligned
        v16bf b0, b1;
        #pragma unroll
        for (int i = 0; i < 8; ++i) {
            b0[i] = bp[i];      b0[8 + i] = bp[16 + i];
            b1[i] = bp[32 + i]; b1[8 + i] = bp[48 + i];
        }
        const float se = scen4[code];
        const float ms = msc2[code];

        v8f acc0 = __builtin_amdgcn_wmma_f32_16x16x32_bf16(false, a00, false, b0,
                                                           (short)0, c0, false, false);
        acc0 = __builtin_amdgcn_wmma_f32_16x16x32_bf16(false, a01, false, b1,
                                                       (short)0, acc0, false, false);
        v8f acc1 = __builtin_amdgcn_wmma_f32_16x16x32_bf16(false, a10, false, b0,
                                                           (short)0, c1, false, false);
        acc1 = __builtin_amdgcn_wmma_f32_16x16x32_bf16(false, a11, false, b1,
                                                       (short)0, acc1, false, false);

        #pragma unroll
        for (int j = 0; j < 8; ++j) {
            float d0 = fmaf(ms, acc0[j], se);          // > 0 by construction
            float d1 = fmaf(ms, acc1[j], se);
            unsigned k0 = (__float_as_uint(d0) & 0xFFFFFE00u) | (unsigned)code;
            unsigned k1 = (__float_as_uint(d1) & 0xFFFFFE00u) | (unsigned)code;
            if (k0 < key0[j]) key0[j] = k0;            // v_min_u32
            if (k1 < key1[j]) key1[j] = k1;
        }
    }

    // ---- per-row argmin across 16 lanes of each half: pure u32-min butterfly ----
    #pragma unroll
    for (int j = 0; j < 8; ++j) {
        #pragma unroll
        for (int s = 1; s < 16; s <<= 1) {
            unsigned o0 = __shfl_xor(key0[j], s, 32);
            unsigned o1 = __shfl_xor(key1[j], s, 32);
            if (o0 < key0[j]) key0[j] = o0;
            if (o1 < key1[j]) key1[j] = o1;
        }
    }
    if (l16 == 0) {
        #pragma unroll
        for (int j = 0; j < 8; ++j) {
            sIdx[wave][j + 8*hi]      = (int)(key0[j] & 0x1FFu);
            sIdx[wave][16 + j + 8*hi] = (int)(key1[j] & 0x1FFu);
        }
    }
    __syncthreads();

    const int qi0 = sIdx[wave][l16];
    const int qi1 = sIdx[wave][16 + l16];
    if (hi == 0) { idxOut[row0] = qi0; idxOut[row1] = qi1; }

    // ---- gather codebook rows, NT-store quantized, accumulate loss ----
    const float* ep0 = emb + qi0 * DIM;
    const float* ep1 = emb + qi1 * DIM;
    float* qp0 = qout + row0 * DIM;
    float* qp1 = qout + row1 * DIM;
    float lsum = 0.0f;
    #pragma unroll
    for (int s2 = 0; s2 < 4; ++s2) {
        const int b = segB[s2];
        f4 q00 = *(const f4*)(ep0 + b);
        f4 q01 = *(const f4*)(ep0 + b + 4);
        f4 q10 = *(const f4*)(ep1 + b);
        f4 q11 = *(const f4*)(ep1 + b + 4);
        f4 d;
        d = q00 - xv0[2*s2];   lsum += d.x*d.x + d.y*d.y + d.z*d.z + d.w*d.w;
        d = q01 - xv0[2*s2+1]; lsum += d.x*d.x + d.y*d.y + d.z*d.z + d.w*d.w;
        d = q10 - xv1[2*s2];   lsum += d.x*d.x + d.y*d.y + d.z*d.z + d.w*d.w;
        d = q11 - xv1[2*s2+1]; lsum += d.x*d.x + d.y*d.y + d.z*d.z + d.w*d.w;
        __builtin_nontemporal_store(q00, (f4*)(qp0 + b));
        __builtin_nontemporal_store(q01, (f4*)(qp0 + b + 4));
        __builtin_nontemporal_store(q10, (f4*)(qp1 + b));
        __builtin_nontemporal_store(q11, (f4*)(qp1 + b + 4));
    }

    // deterministic loss: wave shuffle reduce -> LDS -> ordered block sum
    #pragma unroll
    for (int s = 16; s >= 1; s >>= 1)
        lsum += __shfl_xor(lsum, s, 32);
    if (lane == 0) sLoss[wave] = lsum;
    __syncthreads();
    if (tid == 0) {
        float tot = 0.0f;
        #pragma unroll
        for (int w = 0; w < 8; ++w) tot += sLoss[w];
        partials[blockIdx.x] = tot;
    }
}

// ---- deterministic final loss reduction over BLOCKS partials ----
__global__ __launch_bounds__(256) void vq_loss_kernel(const float* __restrict__ partials,
                                                      float* __restrict__ lossOut) {
    __shared__ float buf[256];
    const int t = threadIdx.x;
    float s = 0.0f;
    #pragma unroll
    for (int i = 0; i < BLOCKS / 256; ++i) s += partials[t * (BLOCKS / 256) + i];
    buf[t] = s;
    __syncthreads();
    #pragma unroll
    for (int k = 128; k > 0; k >>= 1) {
        if (t < k) buf[t] += buf[t + k];
        __syncthreads();
    }
    if (t == 0) *lossOut = buf[0] * (0.25f / 16777216.0f);
}

extern "C" void kernel_launch(void* const* d_in, const int* in_sizes, int n_in,
                              void* d_out, int out_size, void* d_ws, size_t ws_size,
                              hipStream_t stream) {
    const float* x   = (const float*)d_in[0];   // [64,4096,64]
    const float* emb = (const float*)d_in[1];   // [512,64]
    const float* ema = (const float*)d_in[2];   // [512]

    __bf16* embB     = (__bf16*)d_ws;                                   // 64 KB
    float*  scen4    = (float*)((char*)d_ws + 65536);                   // 2 KB
    float*  msc2     = (float*)((char*)d_ws + 65536 + 2048);            // 2 KB
    float*  partials = (float*)((char*)d_ws + 65536 + 4096);            // 4 KB (1024)

    float* out    = (float*)d_out;
    float* qout   = out;
    float* lossP  = out + QELEMS;
    int*   idxOut = (int*)(out + QELEMS + 1);

    vq_prep_kernel<<<2, 256, 0, stream>>>(emb, ema, embB, scen4, msc2);
    vq_main_kernel<<<BLOCKS, 256, 0, stream>>>(x, emb, embB, scen4, msc2,
                                               qout, idxOut, partials);
    vq_loss_kernel<<<1, 256, 0, stream>>>(partials, lossP);
}
